// GNNModel_84301618086559
// MI455X (gfx1250) — matrix operand order repaired and checked by reading, not compiled, not run on hardware
//
#include <hip/hip_runtime.h>
#include <hip/hip_bf16.h>

typedef __attribute__((ext_vector_type(16))) _Float16 v16h;
typedef __attribute__((ext_vector_type(8)))  float    v8f;

#define HEADS 4
#define CDIM 16
#define FDIM 64   // HEADS*CDIM

// ---------- helpers ----------
__device__ __forceinline__ unsigned f2key(float f) {
  unsigned b = __float_as_uint(f);
  return (b & 0x80000000u) ? ~b : (b | 0x80000000u);
}
__device__ __forceinline__ float key2f(unsigned u) {
  unsigned b = (u & 0x80000000u) ? (u ^ 0x80000000u) : ~u;
  return __uint_as_float(b);
}

// ---------- 0. edge list: int32 + self loops ----------
__global__ __launch_bounds__(256) void build_edges(const int* __restrict__ ei,
                                                   int* __restrict__ srcI,
                                                   int* __restrict__ dstI,
                                                   long long nE, long long nEt) {
  long long t = (long long)blockIdx.x * 256 + threadIdx.x;
  if (t >= nEt) return;
  if (t < nE) { srcI[t] = ei[t]; dstI[t] = ei[nE + t]; }
  else        { int v = (int)(t - nE); srcI[t] = v; dstI[t] = v; }
}

// ---------- 1. resets ----------
__global__ __launch_bounds__(256) void reset_layer(float* __restrict__ agg,
                                                   float* __restrict__ denom,
                                                   unsigned* __restrict__ mkey,
                                                   int n) {
  int t = blockIdx.x * 256 + threadIdx.x;
  if (t < n * FDIM) agg[t] = 0.0f;
  if (t < n * HEADS) { denom[t] = 0.0f; mkey[t] = 0x007FFFFFu; /* key(-inf) */ }
}

__global__ __launch_bounds__(256) void reset_pool(float* __restrict__ pooled,
                                                  float* __restrict__ counts,
                                                  int g) {
  int t = blockIdx.x * 256 + threadIdx.x;
  if (t < g * FDIM) pooled[t] = 0.0f;
  if (t < g) counts[t] = 0.0f;
}

// ---------- 2. WMMA GEMM: out[n,64] = A[n,K] @ W[K,64] ----------
// Block = 4 waves, each wave one 16-row tile. W is cooperatively staged into
// LDS once per block, pre-swizzled into per-lane B fragments so each wave's
// B feed is a single contiguous 32-byte ds_load per tile.
// A layout (16-bit A 16x32, ISA 7.12.2): lane L holds row M=L%16,
//   half i in 0..7  -> k = (L/16)*8 + i        (contiguous, float4 x2)
//   half i in 8..15 -> k = (L/16)*8 + 16+(i-8) (contiguous, float4 x2)
// B layout (32x16): lane L holds col N=L%16, half i -> k = (L/16)*8 + i (+8 if i>=8), same as A
__global__ __launch_bounds__(128) void gemm_wmma(const float* __restrict__ A,
                                                 const float* __restrict__ W,
                                                 float* __restrict__ out,
                                                 int n, int K) {
  __shared__ _Float16 Bsh[4096];   // up to K=64: 2 chunks * 4 tiles * 32 lanes * 16 halves

  const int wave = threadIdx.x >> 5;
  const int lane = threadIdx.x & 31;
  const int tile = blockIdx.x * 4 + wave;
  const int lhalf = lane >> 4;
  const int lmod  = lane & 15;
  const int nchunks = K >> 5;

  // --- cooperative stage of W into B-fragment layout (coalesced global reads) ---
  for (int u = threadIdx.x; u < K * FDIM; u += 128) {
    int k = u >> 6, col = u & 63;
    int kt = k >> 5, kk = k & 31;
    int ln = ((kk >> 3) & 1) * 16 + (col & 15);
    int i  = (kk & 7) + ((kk >> 4) << 3);
    int nt = col >> 4;
    Bsh[(((kt << 2) + nt) * 32 + ln) * 16 + i] = (_Float16)W[u];
  }
  __syncthreads();

  if (tile * 16 >= n) return;          // wave-uniform: EXEC all-ones for WMMA

  v8f acc[4] = {{}, {}, {}, {}};
  for (int kt = 0; kt < nchunks; ++kt) {
    const int row = tile * 16 + lmod;
    const float* arow = A + (size_t)row * K + kt * 32 + lhalf * 8;
    const float4 a0 = *(const float4*)(arow + 0);   // k base+0..3
    const float4 a1 = *(const float4*)(arow + 4);   // k base+4..7
    const float4 a2 = *(const float4*)(arow + 16);  // k base+16..19
    const float4 a3 = *(const float4*)(arow + 20);  // k base+20..23
    v16h a;
    a[0]  = (_Float16)a0.x; a[1]  = (_Float16)a0.y; a[2]  = (_Float16)a0.z; a[3]  = (_Float16)a0.w;
    a[4]  = (_Float16)a1.x; a[5]  = (_Float16)a1.y; a[6]  = (_Float16)a1.z; a[7]  = (_Float16)a1.w;
    a[8]  = (_Float16)a2.x; a[9]  = (_Float16)a2.y; a[10] = (_Float16)a2.z; a[11] = (_Float16)a2.w;
    a[12] = (_Float16)a3.x; a[13] = (_Float16)a3.y; a[14] = (_Float16)a3.z; a[15] = (_Float16)a3.w;
#pragma unroll
    for (int nt = 0; nt < 4; ++nt) {
      v16h b = *(const v16h*)&Bsh[(((kt << 2) + nt) * 32 + lane) * 16];
      acc[nt] = __builtin_amdgcn_wmma_f32_16x16x32_f16(
          false, a, false, b, (short)0, acc[nt], false, false);
    }
  }
  // D layout: VGPR r, lanes 0-15 -> M=r ; lanes 16-31 -> M=8+r ; N = lane%16
  const int rowbase = tile * 16 + lhalf * 8;
#pragma unroll
  for (int r = 0; r < 8; ++r) {
    int row = rowbase + r;
    if (row < n) {
      float* orow = out + (size_t)row * FDIM + lmod;
      orow[0]  = acc[0][r];
      orow[16] = acc[1][r];
      orow[32] = acc[2][r];
      orow[48] = acc[3][r];
    }
  }
}

// ---------- 3. attention coefficients: asrc/adst [n, H] ----------
__global__ __launch_bounds__(256) void att_coef(const float* __restrict__ feat,
                                                const float* __restrict__ a_s,
                                                const float* __restrict__ a_d,
                                                float* __restrict__ asrc,
                                                float* __restrict__ adst, int n) {
  int t = blockIdx.x * 256 + threadIdx.x;
  if (t >= n * HEADS) return;
  int node = t >> 2, h = t & 3;
  const float4* fp = (const float4*)(feat + (size_t)node * FDIM + h * CDIM);
  const float4* as = (const float4*)(a_s + h * CDIM);
  const float4* ad = (const float4*)(a_d + h * CDIM);
  float s1 = 0.f, s2 = 0.f;
#pragma unroll
  for (int q = 0; q < 4; ++q) {
    float4 v = fp[q], u = as[q], w = ad[q];
    s1 += v.x * u.x + v.y * u.y + v.z * u.z + v.w * u.w;
    s2 += v.x * w.x + v.y * w.y + v.z * w.z + v.w * w.w;
  }
  asrc[t] = s1; adst[t] = s2;
}

// ---------- 4. edge pass A: segment max of e over dst ----------
__global__ __launch_bounds__(256) void edge_max(const int* __restrict__ src,
                                                const int* __restrict__ dst,
                                                const float* __restrict__ asrc,
                                                const float* __restrict__ adst,
                                                unsigned* __restrict__ mkey,
                                                long long nEt) {
  long long t = (long long)blockIdx.x * 256 + threadIdx.x;
  long long e = t >> 2;
  if (e >= nEt) return;
  int h = (int)(t & 3);
  __builtin_prefetch(src + e + 4096, 0, 0);   // global_prefetch_b8 on edge stream
  int s = src[e], d = dst[e];
  float ev = asrc[(size_t)s * HEADS + h] + adst[(size_t)d * HEADS + h];
  ev = ev > 0.f ? ev : 0.2f * ev;             // leaky_relu(0.2)
  atomicMax(&mkey[(size_t)d * HEADS + h], f2key(ev));
}

// ---------- 5. edge pass B: denom += ex ; agg[dst] += ex * h[src] ----------
// 16 threads per edge, each owning 4 channels: float4 (b128) feature gather,
// softmax normalization folded into the node-wise epilogue.
__global__ __launch_bounds__(256) void edge_agg(const int* __restrict__ src,
                                                const int* __restrict__ dst,
                                                const float* __restrict__ asrc,
                                                const float* __restrict__ adst,
                                                const unsigned* __restrict__ mkey,
                                                const float* __restrict__ feat,
                                                float* __restrict__ agg,
                                                float* __restrict__ denom,
                                                long long nEt) {
  long long t = (long long)blockIdx.x * 256 + threadIdx.x;
  long long e = t >> 4;
  if (e >= nEt) return;
  int q = (int)(t & 15);        // 4-channel group
  int c0 = q << 2;
  int h = q >> 2;
  __builtin_prefetch(dst + e + 2048, 0, 0);
  int s = src[e], d = dst[e];
  float ev = asrc[(size_t)s * HEADS + h] + adst[(size_t)d * HEADS + h];
  ev = ev > 0.f ? ev : 0.2f * ev;
  float mx = key2f(mkey[(size_t)d * HEADS + h]);
  float ex = __expf(ev - mx);
  if ((q & 3) == 0) atomicAdd(&denom[(size_t)d * HEADS + h], ex);
  const float4 f = *(const float4*)&feat[(size_t)s * FDIM + c0];
  float* ap = &agg[(size_t)d * FDIM + c0];
  atomicAdd(ap + 0, ex * f.x);
  atomicAdd(ap + 1, ex * f.y);
  atomicAdd(ap + 2, ex * f.z);
  atomicAdd(ap + 3, ex * f.w);
}

// ---------- 6. normalize + bias + ELU (b128 in/out) ----------
__global__ __launch_bounds__(256) void norm_elu(const float* __restrict__ agg,
                                                const float* __restrict__ denom,
                                                const float* __restrict__ bias,
                                                float* __restrict__ outfeat, int n) {
  int t = blockIdx.x * 256 + threadIdx.x;
  if (t >= n * CDIM) return;        // n*64/4 work items
  int node = t >> 4, q = t & 15;
  int c0 = q << 2;
  float dinv = 1.0f / denom[(size_t)node * HEADS + (q >> 2)];
  float4 v = *(const float4*)&agg[(size_t)node * FDIM + c0];
  float4 b = *(const float4*)&bias[c0];
  v.x = v.x * dinv + b.x; v.y = v.y * dinv + b.y;
  v.z = v.z * dinv + b.z; v.w = v.w * dinv + b.w;
  v.x = v.x > 0.f ? v.x : (__expf(v.x) - 1.f);
  v.y = v.y > 0.f ? v.y : (__expf(v.y) - 1.f);
  v.z = v.z > 0.f ? v.z : (__expf(v.z) - 1.f);
  v.w = v.w > 0.f ? v.w : (__expf(v.w) - 1.f);
  *(float4*)&outfeat[(size_t)node * FDIM + c0] = v;
}

// ---------- 7. global mean pool (sum + counts) ----------
__global__ __launch_bounds__(256) void pool_sum(const float* __restrict__ feat,
                                                const int* __restrict__ batch,
                                                float* __restrict__ pooled,
                                                float* __restrict__ counts, int n) {
  int t = blockIdx.x * 256 + threadIdx.x;
  if (t >= n * CDIM) return;        // 16 threads/node, 4 channels each
  int node = t >> 4, q = t & 15;
  int c0 = q << 2;
  int g = batch[node];
  float4 f = *(const float4*)&feat[(size_t)node * FDIM + c0];
  float* pp = &pooled[(size_t)g * FDIM + c0];
  atomicAdd(pp + 0, f.x);
  atomicAdd(pp + 1, f.y);
  atomicAdd(pp + 2, f.z);
  atomicAdd(pp + 3, f.w);
  if (q == 0) atomicAdd(&counts[g], 1.0f);
}

// ---------- 8. fusion head ----------
__global__ __launch_bounds__(256) void head_kernel(const float* __restrict__ pooled,
                                                   const float* __restrict__ counts,
                                                   const float* __restrict__ guidance,
                                                   const float* __restrict__ Wg,
                                                   const float* __restrict__ bg,
                                                   const float* __restrict__ Wu,
                                                   const float* __restrict__ bu,
                                                   const float* __restrict__ Wo,
                                                   const float* __restrict__ bo,
                                                   float* __restrict__ out, int gtot) {
  int g = blockIdx.x * 256 + threadIdx.x;
  if (g >= gtot) return;
  float inv = 1.0f / fmaxf(counts[g], 1.0f);
  float gu = guidance[g];
  float tcol[CDIM];
#pragma unroll
  for (int c = 0; c < CDIM; ++c) {
    float xg = 0.f;
    for (int k = 0; k < FDIM; ++k) xg += pooled[(size_t)g * FDIM + k] * Wg[k * CDIM + c];
    xg = xg * inv + bg[c];
    float xu = fmaxf(gu * Wu[c] + bu[c], 0.0f);
    tcol[c] = xg + xu;
  }
#pragma unroll
  for (int j = 0; j < 7; ++j) {
    float o = bo[j];
#pragma unroll
    for (int c = 0; c < CDIM; ++c) o += tcol[c] * Wo[c * 7 + j];
    out[(size_t)g * 7 + j] = o;
  }
}

// ===================== host side =====================
extern "C" void kernel_launch(void* const* d_in, const int* in_sizes, int n_in,
                              void* d_out, int out_size, void* d_ws, size_t ws_size,
                              hipStream_t stream) {
  (void)n_in; (void)out_size; (void)ws_size;

  const float* x        = (const float*)d_in[0];
  const int*   eidx     = (const int*)d_in[1];
  const int*   batch    = (const int*)d_in[2];
  const float* guidance = (const float*)d_in[3];
  const float* W1    = (const float*)d_in[4];
  const float* a_s1  = (const float*)d_in[5];
  const float* a_d1  = (const float*)d_in[6];
  const float* b1    = (const float*)d_in[7];
  const float* W2    = (const float*)d_in[8];
  const float* a_s2  = (const float*)d_in[9];
  const float* a_d2  = (const float*)d_in[10];
  const float* b2    = (const float*)d_in[11];
  const float* Wg    = (const float*)d_in[12];
  const float* bg    = (const float*)d_in[13];
  const float* Wu    = (const float*)d_in[14];
  const float* bu    = (const float*)d_in[15];
  const float* Wo    = (const float*)d_in[16];
  const float* bo    = (const float*)d_in[17];
  float* out = (float*)d_out;

  const long long N  = in_sizes[0] / 32;        // DIN = 32
  const long long E  = in_sizes[1] / 2;
  const long long G  = in_sizes[3];
  const long long Et = E + N;

  // carve workspace
  char* ws = (char*)d_ws;
  size_t off = 0;
  auto carve = [&](size_t bytes) -> void* {
    void* p = ws + off;
    off = (off + bytes + 255) & ~(size_t)255;
    return p;
  };
  int*      srcI   = (int*)carve(Et * sizeof(int));
  int*      dstI   = (int*)carve(Et * sizeof(int));
  float*    featA  = (float*)carve((size_t)N * FDIM * sizeof(float)); // layer outputs
  float*    featB  = (float*)carve((size_t)N * FDIM * sizeof(float)); // gemm output h
  float*    agg    = (float*)carve((size_t)N * FDIM * sizeof(float));
  float*    asrc   = (float*)carve((size_t)N * HEADS * sizeof(float));
  float*    adst   = (float*)carve((size_t)N * HEADS * sizeof(float));
  unsigned* mkey   = (unsigned*)carve((size_t)N * HEADS * sizeof(unsigned));
  float*    denom  = (float*)carve((size_t)N * HEADS * sizeof(float));
  float*    pooled = (float*)carve((size_t)G * FDIM * sizeof(float));
  float*    counts = (float*)carve((size_t)G * sizeof(float));

  auto cdiv = [](long long a, long long b) { return (unsigned)((a + b - 1) / b); };
  const unsigned gNodeF  = cdiv(N * FDIM, 256);
  const unsigned gNodeV4 = cdiv(N * CDIM, 256);   // 16 threads/node (float4 kernels)
  const unsigned gNodeH  = cdiv(N * HEADS, 256);
  const unsigned gEdge4  = cdiv(Et * 4, 256);
  const unsigned gEdge16 = cdiv(Et * 16, 256);
  const unsigned gGemm   = cdiv((N + 15) / 16, 4); // 4 waves/block
  const unsigned gPool   = cdiv(G * FDIM, 256);

  // edge preprocessing
  build_edges<<<cdiv(Et, 256), 256, 0, stream>>>(eidx, srcI, dstI, E, Et);

  // -------- layer 1 --------
  reset_layer<<<gNodeF, 256, 0, stream>>>(agg, denom, mkey, (int)N);
  gemm_wmma<<<gGemm, 128, 0, stream>>>(x, W1, featB, (int)N, 32);
  att_coef<<<gNodeH, 256, 0, stream>>>(featB, a_s1, a_d1, asrc, adst, (int)N);
  edge_max<<<gEdge4, 256, 0, stream>>>(srcI, dstI, asrc, adst, mkey, Et);
  edge_agg<<<gEdge16, 256, 0, stream>>>(srcI, dstI, asrc, adst, mkey, featB, agg, denom, Et);
  norm_elu<<<gNodeV4, 256, 0, stream>>>(agg, denom, b1, featA, (int)N);

  // -------- layer 2 --------
  reset_layer<<<gNodeF, 256, 0, stream>>>(agg, denom, mkey, (int)N);
  gemm_wmma<<<gGemm, 128, 0, stream>>>(featA, W2, featB, (int)N, 64);
  att_coef<<<gNodeH, 256, 0, stream>>>(featB, a_s2, a_d2, asrc, adst, (int)N);
  edge_max<<<gEdge4, 256, 0, stream>>>(srcI, dstI, asrc, adst, mkey, Et);
  edge_agg<<<gEdge16, 256, 0, stream>>>(srcI, dstI, asrc, adst, mkey, featB, agg, denom, Et);
  norm_elu<<<gNodeV4, 256, 0, stream>>>(agg, denom, b2, featA, (int)N);

  // -------- pool + head --------
  reset_pool<<<gPool, 256, 0, stream>>>(pooled, counts, (int)G);
  pool_sum<<<gNodeV4, 256, 0, stream>>>(featA, batch, pooled, counts, (int)N);
  head_kernel<<<cdiv(G, 256), 256, 0, stream>>>(pooled, counts, guidance,
                                                Wg, bg, Wu, bu, Wo, bo, out, (int)G);
}
